// ScaledDotProductAttention_31233002176653
// MI455X (gfx1250) — matrix operand order repaired and checked by reading, not compiled, not run on hardware
//
#include <hip/hip_runtime.h>
#include <math.h>

typedef __attribute__((ext_vector_type(16))) _Float16 v16h;
typedef __attribute__((ext_vector_type(8)))  _Float16 v8h;
typedef __attribute__((ext_vector_type(8)))  float    v8f;

#define B_       16
#define L_       2048
#define D_       64
#define NW_      16          // waves per block (512 threads, 4 waves/SIMD)
#define TPW_     8           // k-tiles per wave in the QK phase (16*8 = 128)
#define INV_TEMP 0.125f
#define EPS_     1e-7f
#define NEG_INF_ (-3.402823466e38f)

#if __has_builtin(__builtin_amdgcn_sched_barrier)
#define SCHED_FENCE() __builtin_amdgcn_sched_barrier(0)
#else
#define SCHED_FENCE() asm volatile("" ::: "memory")
#endif

// merge two sorted-descending 4-lists, keep top-4 in a.
__device__ inline void merge_top4(float* a, const float* b) {
  float out[4];
  int i = 0, j = 0;
  #pragma unroll
  for (int n = 0; n < 4; ++n) {
    const float av = a[i], bv = b[j];
    if (av >= bv) { out[n] = av; ++i; } else { out[n] = bv; ++j; }
  }
  #pragma unroll
  for (int n = 0; n < 4; ++n) a[n] = out[n];
}

// ---------------------------------------------------------------------------
// Fully fused sparse attention: one block per (batch, 16-row q-tile).
// 512 threads = 16 wave32 (4 waves/SIMD -> ~256 VGPR budget, latency hidden
// by multithreading). sched_barrier(0) between k-tiles caps f32->f16 staging
// liveness so scores (64 VGPRs) + stats stay resident without spills.
// w hits HBM exactly once; PV reads the f16 tile from swizzled LDS via WMMA.
// ---------------------------------------------------------------------------
__global__ __launch_bounds__(512) void fused_sparse_attn_kernel(
    const float* __restrict__ q, const float* __restrict__ kmat,
    const float* __restrict__ vmat, float* __restrict__ out,
    float* __restrict__ w)
{
  __shared__ __align__(16) char smem[65536];
  float*    red   = (float*)smem;             // [16 waves][16 rows]  (1 KB)
  float*    t4s   = (float*)(smem + 1024);    // [16 waves][16 rows][4] (4 KB)
  float*    rstat = (float*)(smem + 5120);    // [16 rows] final stat
  _Float16* tile  = (_Float16*)smem;          // [16][2048] f16, swizzled (64 KB)
  float*    part  = (float*)smem;             // [16 waves][32 lanes][8] (16 KB)

  const int tid  = threadIdx.x;
  const int lane = tid & 31;
  const int wv   = tid >> 5;
  const int l15  = lane & 15;
  const int grp  = lane >> 4;
  const int b    = blockIdx.x >> 7;
  const int qt   = blockIdx.x & 127;

  // ===== Phase 1: scores for this wave's 8 k-tiles ==========================
  const float* qrow = q + ((size_t)b * L_ + qt * 16 + l15) * D_;
  v16h a0, a1;                                // A frag (Q/8), reused 8x
  #pragma unroll
  for (int e = 0; e < 16; ++e) {
    const int d = ((e >> 3) << 4) + (grp << 3) + (e & 7);
    a0[e] = (_Float16)(qrow[d]      * INV_TEMP);
    a1[e] = (_Float16)(qrow[32 + d] * INV_TEMP);
  }
  v8f s[TPW_];
  #pragma unroll
  for (int t = 0; t < TPW_; ++t) {
    const int kt = wv * TPW_ + t;
    const float* krow = kmat + ((size_t)b * L_ + (size_t)kt * 16 + l15) * D_;
    v16h b0, b1;
    #pragma unroll
    for (int e = 0; e < 16; ++e) {
      const int d = ((e >> 3) << 4) + (grp << 3) + (e & 7);
      b0[e] = (_Float16)krow[d];
      b1[e] = (_Float16)krow[32 + d];
    }
    v8f acc = {};
    acc = __builtin_amdgcn_wmma_f32_16x16x32_f16(false, a0, false, b0,
                                                 (short)0, acc, false, false);
    acc = __builtin_amdgcn_wmma_f32_16x16x32_f16(false, a1, false, b1,
                                                 (short)0, acc, false, false);
    s[t] = acc;
    SCHED_FENCE();   // cap staging-register liveness to one k-tile
  }

  // ===== Phase 2a: row max (wave shfl -> red, 16 row-threads -> rstat) =====
  float pm[8];
  #pragma unroll
  for (int r = 0; r < 8; ++r) {
    float m = NEG_INF_;
    #pragma unroll
    for (int t = 0; t < TPW_; ++t) m = fmaxf(m, s[t][r]);
    pm[r] = m;
  }
  #pragma unroll
  for (int o = 8; o > 0; o >>= 1) {
    #pragma unroll
    for (int r = 0; r < 8; ++r) pm[r] = fmaxf(pm[r], __shfl_xor(pm[r], o, 32));
  }
  if (l15 == 0) {
    #pragma unroll
    for (int r = 0; r < 8; ++r) red[wv * 16 + grp * 8 + r] = pm[r];
  }
  __syncthreads();
  if (tid < 16) {
    float m = red[tid];
    for (int w2 = 1; w2 < NW_; ++w2) m = fmaxf(m, red[w2 * 16 + tid]);
    rstat[tid] = m;
  }
  __syncthreads();
  float rmax[8];
  #pragma unroll
  for (int r = 0; r < 8; ++r) rmax[r] = rstat[grp * 8 + r];

  // ===== Phase 2b: exp + sum -> probabilities ==============================
  float ps[8] = {0, 0, 0, 0, 0, 0, 0, 0};
  #pragma unroll
  for (int t = 0; t < TPW_; ++t) {
    #pragma unroll
    for (int r = 0; r < 8; ++r) {
      const float e = __expf(s[t][r] - rmax[r]);
      s[t][r] = e;
      ps[r] += e;
    }
  }
  #pragma unroll
  for (int o = 8; o > 0; o >>= 1) {
    #pragma unroll
    for (int r = 0; r < 8; ++r) ps[r] += __shfl_xor(ps[r], o, 32);
  }
  if (l15 == 0) {
    #pragma unroll
    for (int r = 0; r < 8; ++r) red[wv * 16 + grp * 8 + r] = ps[r];
  }
  __syncthreads();
  if (tid < 16) {
    float ss = red[tid];
    for (int w2 = 1; w2 < NW_; ++w2) ss += red[w2 * 16 + tid];
    rstat[tid] = ss;
  }
  __syncthreads();
  float inv[8];
  #pragma unroll
  for (int r = 0; r < 8; ++r) inv[r] = 1.0f / rstat[grp * 8 + r];
  #pragma unroll
  for (int t = 0; t < TPW_; ++t)
    #pragma unroll
    for (int r = 0; r < 8; ++r) s[t][r] *= inv[r];

  // ===== Phase 2c: top-4 per row ===========================================
  float t4r[8][4];
  #pragma unroll
  for (int r = 0; r < 8; ++r) {
    t4r[r][0] = t4r[r][1] = t4r[r][2] = t4r[r][3] = -1.0f;  // probs >= 0
    #pragma unroll
    for (int t = 0; t < TPW_; ++t) {
      const float v = s[t][r];
      if (v > t4r[r][3]) {
        t4r[r][3] = v;
        #pragma unroll
        for (int p = 3; p > 0; --p) {
          if (t4r[r][p] > t4r[r][p - 1]) {
            const float tmp = t4r[r][p];
            t4r[r][p] = t4r[r][p - 1];
            t4r[r][p - 1] = tmp;
          }
        }
      }
    }
  }
  #pragma unroll
  for (int o = 8; o > 0; o >>= 1) {
    #pragma unroll
    for (int r = 0; r < 8; ++r) {
      float ot[4];
      #pragma unroll
      for (int j = 0; j < 4; ++j) ot[j] = __shfl_xor(t4r[r][j], o, 32);
      merge_top4(t4r[r], ot);
    }
  }
  if (l15 == 0) {
    #pragma unroll
    for (int r = 0; r < 8; ++r)
      #pragma unroll
      for (int j = 0; j < 4; ++j)
        t4s[(wv * 16 + grp * 8 + r) * 4 + j] = t4r[r][j];
  }
  __syncthreads();
  if (tid < 16) {                              // one thread per row
    float g[4] = { t4s[tid * 4 + 0], t4s[tid * 4 + 1],
                   t4s[tid * 4 + 2], t4s[tid * 4 + 3] };
    for (int w2 = 1; w2 < NW_; ++w2) merge_top4(g, &t4s[(w2 * 16 + tid) * 4]);
    rstat[tid] = g[3] + EPS_;                  // delta = 4th-largest + eps
  }
  __syncthreads();
  float dl_[8];
  #pragma unroll
  for (int r = 0; r < 8; ++r) dl_[r] = rstat[grp * 8 + r];

  // ===== Phase 2d: threshold + renormalize =================================
  float ps2[8] = {0, 0, 0, 0, 0, 0, 0, 0};
  #pragma unroll
  for (int t = 0; t < TPW_; ++t) {
    #pragma unroll
    for (int r = 0; r < 8; ++r) {
      const float x = fmaxf(s[t][r] - dl_[r], 0.0f);
      s[t][r] = x;
      ps2[r] += x;
    }
  }
  #pragma unroll
  for (int o = 8; o > 0; o >>= 1) {
    #pragma unroll
    for (int r = 0; r < 8; ++r) ps2[r] += __shfl_xor(ps2[r], o, 32);
  }
  if (l15 == 0) {
    #pragma unroll
    for (int r = 0; r < 8; ++r) red[wv * 16 + grp * 8 + r] = ps2[r];
  }
  __syncthreads();
  if (tid < 16) {
    float ss = red[tid];
    for (int w2 = 1; w2 < NW_; ++w2) ss += red[w2 * 16 + tid];
    rstat[tid] = ss;
  }
  __syncthreads();
  float inv2[8];
  #pragma unroll
  for (int r = 0; r < 8; ++r) inv2[r] = 1.0f / (rstat[grp * 8 + r] + EPS_);
  __syncthreads();   // scratch reads done; tile may now overwrite smem

  // ===== write w to HBM (once) + f16 tile to LDS (chunk^row swizzle) =======
  float* wbase = w + ((size_t)b * L_ + (size_t)qt * 16) * L_;
  #pragma unroll
  for (int t = 0; t < TPW_; ++t) {
    const int kt    = wv * TPW_ + t;
    const int col   = kt * 16 + l15;
    const int chunk = col >> 3;
    #pragma unroll
    for (int r = 0; r < 8; ++r) {
      const int m = grp * 8 + r;
      const float val = s[t][r] * inv2[r];
      wbase[(size_t)m * L_ + col] = val;
      tile[m * 2048 + ((chunk ^ m) << 3) + (col & 7)] = (_Float16)val;
    }
    SCHED_FENCE();
  }
  __syncthreads();

  // ===== Phase 3: out = w @ V; 16 waves = 4 d-tiles x 4 k-chunks of 512 ====
  const int dt = wv & 3;                       // Dv tile (16 wide)
  const int kc = wv >> 2;                      // 512-key chunk
  const float* vbase = vmat + (size_t)b * L_ * D_ + dt * 16 + l15;
  const _Float16* trow = tile + (size_t)l15 * 2048;   // A row = lane&15

  v8f acc = {};
  for (int kt = kc * 512; kt < kc * 512 + 512; kt += 32) {
    if (kt + 32 < L_)
      __builtin_prefetch(vbase + (size_t)(kt + 32) * D_, 0, 1);

    const int c0 = (kt >> 3) + grp;            // 16B chunk index of K-block
    const v8h lo = *(const v8h*)(trow + (((c0    ) ^ l15) << 3));
    const v8h hi = *(const v8h*)(trow + (((c0 + 2) ^ l15) << 3));
    v16h a;
    #pragma unroll
    for (int j = 0; j < 8; ++j) { a[j] = lo[j]; a[8 + j] = hi[j]; }

    v16h bm;
    #pragma unroll
    for (int e = 0; e < 16; ++e)
      bm[e] = (_Float16)vbase[(size_t)(kt + ((e >> 3) << 4) + (grp << 3) + (e & 7)) * D_];

    acc = __builtin_amdgcn_wmma_f32_16x16x32_f16(false, a, false, bm,
                                                 (short)0, acc, false, false);
    SCHED_FENCE();   // cap V staging liveness to one 32-key step
  }
  __syncthreads();   // all tile reads done; partials may overwrite smem

  float* pb = part + (size_t)(wv * 32 + lane) * 8;
  #pragma unroll
  for (int r = 0; r < 8; ++r) pb[r] = acc[r];
  __syncthreads();

  if (kc == 0) {
    float* orow = out + ((size_t)b * L_ + (size_t)qt * 16) * D_ + dt * 16;
    #pragma unroll
    for (int r = 0; r < 8; ++r) {
      float sum = 0.0f;
      #pragma unroll
      for (int c = 0; c < 4; ++c)
        sum += part[(size_t)((c * 4 + dt) * 32 + lane) * 8 + r];
      orow[(size_t)(grp * 8 + r) * D_ + l15] = sum;
    }
  }
}

// ---------------------------------------------------------------------------
extern "C" void kernel_launch(void* const* d_in, const int* in_sizes, int n_in,
                              void* d_out, int out_size, void* d_ws, size_t ws_size,
                              hipStream_t stream) {
  (void)in_sizes; (void)n_in; (void)out_size; (void)d_ws; (void)ws_size;
  const float* q = (const float*)d_in[0];
  const float* k = (const float*)d_in[1];
  const float* v = (const float*)d_in[2];
  float* out = (float*)d_out;
  float* w   = out + (size_t)B_ * L_ * D_;     // tuple: out first, then w

  fused_sparse_attn_kernel<<<B_ * (L_ / 16), 512, 0, stream>>>(q, k, v, out, w);
}
